// CrossModeAttention_27650999452505
// MI455X (gfx1250) — matrix operand order repaired and checked by reading, not compiled, not run on hardware
//
#include <hip/hip_runtime.h>

typedef __bf16 bf16;
typedef __attribute__((ext_vector_type(16))) __bf16 v16bf;
typedef __attribute__((ext_vector_type(8)))  __bf16 v8bf;
typedef __attribute__((ext_vector_type(8)))  float  v8f;

#define B_     4
#define N_     1024
#define C_     512
#define H_     8
#define DH_    64
#define DTOT_  1024          // D1+D2
#define CBIG_  1536          // Q(512)+K(512)+V(512) fused output cols

__device__ __forceinline__ v8f wmma_bf16(v16bf a, v16bf b, v8f c) {
  return __builtin_amdgcn_wmma_f32_16x16x32_bf16(false, a, false, b, (short)0, c, false, false);
}

// ---------------------------------------------------------------------------
// Kernel 1: fp32 -> bf16 packing of concat(x1,x2) and fused zero-padded weight
// ---------------------------------------------------------------------------
__global__ void pack_kernel(const float* __restrict__ x1, const float* __restrict__ x2,
                            const float* __restrict__ Wq, const float* __restrict__ Wk,
                            const float* __restrict__ Wv,
                            bf16* __restrict__ xcat, bf16* __restrict__ Wbig) {
  int i = blockIdx.x * 256 + threadIdx.x;
  const int NX = B_ * N_ * DTOT_;              // 4,194,304
  if (i < NX) {
    int row = i >> 10, col = i & 1023;         // row in [0,4096)
    float v = (col < 512) ? x1[row * 512 + col] : x2[row * 512 + (col - 512)];
    xcat[i] = (bf16)v;
  } else {
    int j = i - NX;
    if (j < CBIG_ * DTOT_) {
      int r = j >> 10, k = j & 1023;
      float v = 0.f;
      if (r < 512)        { if (k < 512)  v = Wq[r * 512 + k]; }
      else if (r < 1024)  { if (k >= 512) v = Wk[(r - 512) * 512 + (k - 512)]; }
      else                {               v = Wv[(r - 1024) * 1024 + k]; }
      Wbig[j] = (bf16)v;
    }
  }
}

// ---------------------------------------------------------------------------
// Kernel 2: fused QKV projection GEMM  [4096x1024] x [1536x1024]^T
// one wave -> one 16x64 output strip: A fragment reused across 4 WMMAs/K-step
// ---------------------------------------------------------------------------
__global__ void qkv_gemm(const bf16* __restrict__ xcat, const bf16* __restrict__ Wbig,
                         bf16* __restrict__ Qh, bf16* __restrict__ Kh,
                         bf16* __restrict__ Vt, float* __restrict__ Vf32) {
  const int lane = threadIdx.x & 31;
  const int w    = threadIdx.x >> 5;
  const int half = lane >> 4;                  // 0/1: lane group
  const int ln   = lane & 15;
  const int tid  = blockIdx.x * 4 + w;         // 6144 wave-strips total
  const int tr   = tid / (CBIG_ / 64);         // 0..255  (16-row tile)
  const int tcg  = tid % (CBIG_ / 64);         // 0..23   (64-col group)

  const bf16* arow = xcat + (size_t)(tr * 16 + ln) * DTOT_;                // A: m = ln
  const bf16* brow[4];
#pragma unroll
  for (int u = 0; u < 4; u++)
    brow[u] = Wbig + (size_t)(tcg * 64 + u * 16 + ln) * DTOT_;             // B: n = ln

  v8f acc[4] = {{}, {}, {}, {}};
  for (int kb = 0; kb < DTOT_; kb += 32) {
    // A fragment: lane (m=ln), k = e + 16*half
    int ka = kb + half * 16;
    v8bf lo = *(const v8bf*)(arow + ka);
    v8bf hi = *(const v8bf*)(arow + ka + 8);
    v16bf A;
#pragma unroll
    for (int i = 0; i < 8; i++) { A[i] = lo[i]; A[i + 8] = hi[i]; }
    // B fragments: lane (n=ln), k = e + 16*half -> 16 contiguous bf16
#pragma unroll
    for (int u = 0; u < 4; u++) {
      v16bf Bm = *(const v16bf*)(brow[u] + kb + half * 16);
      acc[u] = wmma_bf16(A, Bm, acc[u]);
    }
  }

  // C/D layout: element r -> row m = r + 8*half, col n = ln
#pragma unroll
  for (int u = 0; u < 4; u++) {
    const int c = tcg * 64 + u * 16 + ln;
#pragma unroll
    for (int r = 0; r < 8; r++) {
      int row = tr * 16 + r + half * 8;        // 0..4095
      int b = row >> 10, n = row & 1023;
      float v = acc[u][r];
      if (c < 512) {                                         // Q
        int h = c >> 6, dh = c & 63;
        Qh[(((size_t)(b * 8 + h) * N_) + n) * DH_ + dh] = (bf16)v;
      } else if (c < 1024) {                                 // K
        int c2 = c - 512, h = c2 >> 6, dh = c2 & 63;
        Kh[(((size_t)(b * 8 + h) * N_) + n) * DH_ + dh] = (bf16)v;
      } else {                                               // V
        int cv = c - 1024, h = cv >> 6, dh = cv & 63;
        Vf32[(size_t)row * C_ + cv] = v;
        Vt[(((size_t)(b * 8 + h) * DH_) + dh) * N_ + n] = (bf16)v;
      }
    }
  }
}

// ---------------------------------------------------------------------------
// Kernel 3: flash-style masked attention, one wave = 16 query rows of a head
// scores + PV both via bf16 WMMA; P transposed C/D->A layout through LDS
// ---------------------------------------------------------------------------
__global__ void attn_kernel(const bf16* __restrict__ Qh, const bf16* __restrict__ Kh,
                            const bf16* __restrict__ Vt, float* __restrict__ V1) {
  __shared__ bf16 pbuf[4][16][32];             // per-wave P tile (16 rows x 32 keys)
  const int lane = threadIdx.x & 31;
  const int w    = threadIdx.x >> 5;
  const int half = lane >> 4;
  const int ln   = lane & 15;

  const int bh      = blockIdx.x >> 4;         // 0..31 (b*8+h)
  const int rowBlk  = blockIdx.x & 15;
  const int b       = bh >> 3;
  const int h       = bh & 7;
  const int rowBase = rowBlk * 64 + w * 16;    // query rows [rowBase, rowBase+16)

  const bf16* Qp = Qh + (size_t)bh * N_ * DH_;
  const bf16* Kp = Kh + (size_t)bh * N_ * DH_;
  const bf16* Vp = Vt + (size_t)bh * DH_ * N_;

  // Q A-fragments for K(dh)=0..63 -> two 16x32 fragments, loaded once
  v16bf qa[2];
  {
    const bf16* qrow = Qp + (size_t)(rowBase + ln) * DH_;
#pragma unroll
    for (int t = 0; t < 2; t++) {
      int ka = t * 32 + half * 16;
      v8bf lo = *(const v8bf*)(qrow + ka);
      v8bf hi = *(const v8bf*)(qrow + ka + 8);
#pragma unroll
      for (int i = 0; i < 8; i++) { qa[t][i] = lo[i]; qa[t][i + 8] = hi[i]; }
    }
  }

  v8f o[4] = {{}, {}, {}, {}};                 // O accumulator: 16 x 64
  float rmax[8], rsum[8];
#pragma unroll
  for (int r = 0; r < 8; r++) { rmax[r] = -1e30f; rsum[r] = 0.f; }
  const float scale = 0.044194173824159216f;   // 512^-0.5

  for (int kb = 0; kb < N_; kb += 32) {
    // ---- scores: two 16-key tiles, 4 WMMAs ----
    v8f s0 = {}, s1 = {};
    const bf16* krow0 = Kp + (size_t)(kb + ln) * DH_;
    const bf16* krow1 = Kp + (size_t)(kb + 16 + ln) * DH_;
#pragma unroll
    for (int t = 0; t < 2; t++) {
      v16bf B0 = *(const v16bf*)(krow0 + t * 32 + half * 16);
      s0 = wmma_bf16(qa[t], B0, s0);
      v16bf B1 = *(const v16bf*)(krow1 + t * 32 + half * 16);
      s1 = wmma_bf16(qa[t], B1, s1);
    }

    // ---- online masked softmax (mask: attn != 0, per reference) ----
#pragma unroll
    for (int r = 0; r < 8; r++) {
      float v0 = s0[r] * scale, v1 = s1[r] * scale;
      bool  m0 = (v0 != 0.f),   m1 = (v1 != 0.f);
      float a0 = m0 ? v0 : -1e30f;
      float a1 = m1 ? v1 : -1e30f;
      float tmax = fmaxf(a0, a1);
#pragma unroll
      for (int mm = 1; mm < 16; mm <<= 1) tmax = fmaxf(tmax, __shfl_xor(tmax, mm, 32));
      float nmax = fmaxf(rmax[r], tmax);
      float f    = __expf(rmax[r] - nmax);
      float p0   = m0 ? __expf(a0 - nmax) : 0.f;
      float p1   = m1 ? __expf(a1 - nmax) : 0.f;
      float ts   = p0 + p1;
#pragma unroll
      for (int mm = 1; mm < 16; mm <<= 1) ts += __shfl_xor(ts, mm, 32);
      rsum[r] = rsum[r] * f + ts;
      rmax[r] = nmax;
#pragma unroll
      for (int t = 0; t < 4; t++) o[t][r] *= f;
      // C/D layout -> LDS row-major P tile: row = r + 8*half, cols ln / ln+16
      pbuf[w][r + half * 8][ln]      = (bf16)p0;
      pbuf[w][r + half * 8][ln + 16] = (bf16)p1;
    }

    asm volatile("s_wait_dscnt 0" ::: "memory");

    // ---- reload P as A-fragment (16x32) ----
    v16bf Pa;
    {
      const bf16* prow = &pbuf[w][ln][0];
      int c0 = half * 16;
      v8bf lo = *(const v8bf*)(prow + c0);
      v8bf hi = *(const v8bf*)(prow + c0 + 8);
#pragma unroll
      for (int i = 0; i < 8; i++) { Pa[i] = lo[i]; Pa[i + 8] = hi[i]; }
    }

    // ---- O += P @ V : 4 WMMAs, B-fragments contiguous from Vt ----
#pragma unroll
    for (int t = 0; t < 4; t++) {
      const bf16* vrow = Vp + (size_t)(t * 16 + ln) * N_ + kb + half * 16;
      v16bf Bv = *(const v16bf*)vrow;
      o[t] = wmma_bf16(Pa, Bv, o[t]);
    }
  }

  // ---- normalize + store V1 [B,N,C] ----
#pragma unroll
  for (int r = 0; r < 8; r++) {
    float inv = 1.f / fmaxf(rsum[r], 1e-30f);
    int row = rowBase + r + half * 8;
#pragma unroll
    for (int t = 0; t < 4; t++) {
      int col = h * DH_ + t * 16 + ln;
      V1[((size_t)b * N_ + row) * C_ + col] = o[t][r] * inv;
    }
  }
}

// ---------------------------------------------------------------------------
// Kernel 4: residual + LayerNorm; write next V (f32 + bf16 transposed heads)
// one wave per row of C=512 (16 elements/lane)
// ---------------------------------------------------------------------------
__global__ void ln_kernel(const float* __restrict__ V1, const float* __restrict__ Vin,
                          const float* __restrict__ gamma, const float* __restrict__ beta,
                          float* __restrict__ Vout, bf16* __restrict__ Vt,
                          float* __restrict__ out, int write_out) {
  const int lane = threadIdx.x & 31;
  const int w    = threadIdx.x >> 5;
  const int row  = blockIdx.x * 4 + w;         // 0..4095
  const int b = row >> 10, n = row & 1023;

  const float* p1 = V1  + (size_t)row * C_;
  const float* p2 = Vin + (size_t)row * C_;
  const int c0 = lane * 16;
  float x[16];
  float s = 0.f, sq = 0.f;
#pragma unroll
  for (int i = 0; i < 16; i++) {
    x[i] = p1[c0 + i] + p2[c0 + i];
    s += x[i]; sq += x[i] * x[i];
  }
#pragma unroll
  for (int m = 1; m < 32; m <<= 1) { s += __shfl_xor(s, m, 32); sq += __shfl_xor(sq, m, 32); }
  const float mu  = s * (1.f / 512.f);
  const float var = sq * (1.f / 512.f) - mu * mu;
  const float rs  = rsqrtf(var + 1e-5f);
#pragma unroll
  for (int i = 0; i < 16; i++) {
    int c = c0 + i;
    float y = (x[i] - mu) * rs * gamma[c] + beta[c];
    Vout[(size_t)row * C_ + c] = y;
    Vt[(((size_t)(b * 8 + (c >> 6)) * DH_) + (c & 63)) * N_ + n] = (bf16)y;
    if (write_out) out[(size_t)row * C_ + c] = y;
  }
}

// ---------------------------------------------------------------------------
extern "C" void kernel_launch(void* const* d_in, const int* in_sizes, int n_in,
                              void* d_out, int out_size, void* d_ws, size_t ws_size,
                              hipStream_t stream) {
  const float* x1 = (const float*)d_in[0];
  const float* x2 = (const float*)d_in[1];
  const float* Wq = (const float*)d_in[2];
  const float* Wk = (const float*)d_in[3];
  const float* Wv = (const float*)d_in[4];
  const float* gm = (const float*)d_in[5];
  const float* bt = (const float*)d_in[6];
  float* out = (float*)d_out;

  char* ws = (char*)d_ws;
  // workspace layout (bytes)
  bf16*  xcat = (bf16*)(ws + 0);                        //  8 MiB  [4096][1024]
  bf16*  Wbig = (bf16*)(ws + 8388608);                  //  3 MiB  [1536][1024]
  bf16*  Qh   = (bf16*)(ws + 11534336);                 //  4 MiB  [B,H,N,DH]
  bf16*  Kh   = (bf16*)(ws + 15728640);                 //  4 MiB  [B,H,N,DH]
  bf16*  Vt   = (bf16*)(ws + 19922944);                 //  4 MiB  [B,H,DH,N]
  float* Vf32 = (float*)(ws + 24117248);                //  8 MiB  [B,N,C]
  float* V1   = (float*)(ws + 32505856);                //  8 MiB  [B,N,C]

  // 1) bf16 packing (5,767,168 elements)
  pack_kernel<<<22528, 256, 0, stream>>>(x1, x2, Wq, Wk, Wv, xcat, Wbig);
  // 2) fused QKV projections: 256 row-tiles x 24 col-groups, 4 waves/block
  qkv_gemm<<<1536, 128, 0, stream>>>(xcat, Wbig, Qh, Kh, Vt, Vf32);
  // 3) three attention + layernorm layers
  for (int layer = 0; layer < 3; ++layer) {
    attn_kernel<<<512, 128, 0, stream>>>(Qh, Kh, Vt, V1);
    ln_kernel<<<1024, 128, 0, stream>>>(V1, Vf32, gm, bt, Vf32, Vt, out, layer == 2 ? 1 : 0);
  }
}
